// MultiHeadAttention_53463752900838
// MI455X (gfx1250) — matrix operand order repaired and checked by reading, not compiled, not run on hardware
//
#include <hip/hip_runtime.h>
#include <hip/hip_bf16.h>

typedef __bf16 bf16;
typedef __attribute__((ext_vector_type(16))) __bf16 v16bf;
typedef __attribute__((ext_vector_type(8)))  __bf16 v8bf;
typedef __attribute__((ext_vector_type(8)))  float  v8f;
typedef __attribute__((ext_vector_type(4)))  unsigned int u32x4;
typedef __attribute__((ext_vector_type(8)))  int i32x8;
typedef __attribute__((ext_vector_type(4)))  int i32x4;

#define B_DIM   8
#define C_DIM   1024
#define T_DIM   4096
#define NH      16
#define QKV_CH  3072   // 16 heads * (64 q + 64 k + 64 v)

// ---------------------------------------------------------------------------
// f32 -> bf16 conversion (grid-stride)
// ---------------------------------------------------------------------------
__global__ void cvt_f32_bf16(const float* __restrict__ in, bf16* __restrict__ out, long n) {
    long i = (long)blockIdx.x * blockDim.x + threadIdx.x;
    long stride = (long)gridDim.x * blockDim.x;
    for (; i < n; i += stride) out[i] = (bf16)in[i];
}

// ---------------------------------------------------------------------------
// TDM: DMA a 128-row x 32-col bf16 tile (row stride = K elements) from global
// into LDS at byte offset lds_off. D# per CDNA5 ISA ch.8 (2D tensor,
// groups 2/3 zero). Issued once per wave; tracked with TENSORcnt.
// 6-arg builtin form (clang-23 / therock-10.0 headers).
// ---------------------------------------------------------------------------
__device__ __forceinline__ void tdm_load_a_tile(const bf16* gsrc, unsigned lds_off,
                                                int K, int Mtot)
{
    unsigned long long ga = (unsigned long long)(uintptr_t)gsrc;
    u32x4 g0;
    g0[0] = 1u;                                            // count=1, user descriptor
    g0[1] = lds_off;                                       // lds_addr (bytes)
    g0[2] = (unsigned)(ga & 0xffffffffu);                  // global_addr[31:0]
    g0[3] = (unsigned)((ga >> 32) & 0x1ffffffu)            // global_addr[56:32]
          | 0x80000000u;                                   // type = 2 ("image")
    i32x8 g1;
    g1[0] = 0x10000;                                       // data_size=1 (2B), wg_mask=0
    g1[1] = (int)(((unsigned)K & 0xffffu) << 16);          // tensor_dim0[15:0]
    g1[2] = (int)((((unsigned)K >> 16) & 0xffffu)          // tensor_dim0[31:16]
          | (((unsigned)Mtot & 0xffffu) << 16));           // tensor_dim1[15:0]
    g1[3] = (int)((((unsigned)Mtot >> 16) & 0xffffu)       // tensor_dim1[31:16]
          | (32u << 16));                                  // tile_dim0 = 32 elems (64B)
    g1[4] = 128;                                           // tile_dim1 = 128, tile_dim2 = 0
    g1[5] = (int)(unsigned)K;                              // tensor_dim0_stride[31:0]
    g1[6] = 0;                                             // stride hi / dim1_stride lo
    g1[7] = 0;
    i32x4 gz4 = {0, 0, 0, 0};
    i32x8 gz8 = {0, 0, 0, 0, 0, 0, 0, 0};
    __builtin_amdgcn_tensor_load_to_lds(g0, g1, gz4, gz4, gz8, 0);
}

// ---------------------------------------------------------------------------
// Tiled bf16 WMMA GEMM:  Out[b][m][n] = sum_k A[m][k] * Bm[b][k][n] + bias[m]
// Block tile: 128(M) x 64(N), K-step 32, double-buffered LDS.
// 256 threads = 8 wave32 waves; wave w owns M-strip [w*16, w*16+16) x 64 N.
// A tiles arrive via Tensor Data Mover (wave 0, TENSORcnt); B tiles are
// vector-staged transposed so B fragments are two ds_load_b128 each.
// ---------------------------------------------------------------------------
template <typename TOut>
__global__ __launch_bounds__(256) void gemm_bf16_wmma(
    const bf16*  __restrict__ A,     // [M][K] row-major (shared across batch)
    const bf16*  __restrict__ Bm,    // [batch][K][N]
    const float* __restrict__ bias,  // [M]
    TOut*        __restrict__ Out,   // [batch][M][N]
    int M, int K, int N, long bStrideB, long bStrideOut)
{
    __shared__ __align__(16) bf16 As[2][128][32];   // 16 KB (TDM destination)
    __shared__ __align__(16) bf16 Bt[2][64][32];    //  8 KB, transposed: Bt[buf][n][k]

    const int tid  = threadIdx.x;
    const int lane = tid & 31;
    const int wave = tid >> 5;
    const int m0   = blockIdx.y * 128;
    const int n0   = blockIdx.x * 64;
    const int b    = blockIdx.z;

    const bf16* Bsrc  = Bm + (long)b * bStrideB;
    const bf16* Atile = A + (long)m0 * K;           // row stride K

    v8f acc[4];
#pragma unroll
    for (int n = 0; n < 4; ++n) acc[n] = 0.0f;

    // cooperative B-stage indices
    const int bk = tid >> 3;           // 0..31 (k within tile)
    const int bn = (tid & 7) * 8;      // 0..56 (n within tile)

    const int NKS = K / 32;

    // ---- prologue: stage tile 0 into buffer 0 ----
    {
        const bf16* bp = Bsrc + (long)bk * N + (n0 + bn);
        v8bf bv = *reinterpret_cast<const v8bf*>(bp);
#pragma unroll
        for (int i = 0; i < 8; ++i) Bt[0][bn + i][bk] = bv[i];
        if (wave == 0)
            tdm_load_a_tile(Atile, (unsigned)(uintptr_t)&As[0][0][0], K, M);
    }

    int cur = 0;
    for (int kt = 0; kt < NKS; ++kt) {
        // A-tile DMA for 'cur' must have landed before anyone reads it.
        __builtin_amdgcn_s_wait_tensorcnt(0);
        __syncthreads();   // stage(cur) visible; reads of 'nxt' from kt-1 done

        const int nxt = cur ^ 1;
        if (kt + 1 < NKS) {
            const int k0n = (kt + 1) * 32;
            const bf16* bp = Bsrc + (long)(k0n + bk) * N + (n0 + bn);
            v8bf bv = *reinterpret_cast<const v8bf*>(bp);
#pragma unroll
            for (int i = 0; i < 8; ++i) Bt[nxt][bn + i][bk] = bv[i];
            if (wave == 0)
                tdm_load_a_tile(Atile + k0n, (unsigned)(uintptr_t)&As[nxt][0][0], K, M);
            if (kt + 2 < NKS)
                __builtin_prefetch(bp + 32L * N, 0, 3);    // B tile after next
        }

        // ---- A fragment: lane l holds row (wave*16 + l%16);
        //      lanes 0-15: K 0-7 & 16-23, lanes 16-31: K 8-15 & 24-31 ----
        const int ar   = wave * 16 + (lane & 15);
        const int koff = (lane >> 4) * 8;
        v8bf alo = *reinterpret_cast<const v8bf*>(&As[cur][ar][koff]);
        v8bf ahi = *reinterpret_cast<const v8bf*>(&As[cur][ar][koff + 16]);
        v16bf afrag;
#pragma unroll
        for (int i = 0; i < 8; ++i) { afrag[i] = alo[i]; afrag[8 + i] = ahi[i]; }

        // ---- 4 B fragments + 4 WMMAs ----
#pragma unroll
        for (int n = 0; n < 4; ++n) {
            const int col = n * 16 + (lane & 15);
            const int kb  = (lane >> 4) * 8;    // lanes 0-15: K 0-7 & 16-23, 16-31: K 8-15 & 24-31
            v8bf blo = *reinterpret_cast<const v8bf*>(&Bt[cur][col][kb]);
            v8bf bhi = *reinterpret_cast<const v8bf*>(&Bt[cur][col][kb + 16]);
            v16bf bfrag;
#pragma unroll
            for (int i = 0; i < 8; ++i) { bfrag[i] = blo[i]; bfrag[8 + i] = bhi[i]; }

            acc[n] = __builtin_amdgcn_wmma_f32_16x16x32_bf16(
                /*neg_a=*/false, afrag, /*neg_b=*/false, bfrag,
                /*c_mod=*/(short)0, acc[n], /*reuse_a=*/false, /*reuse_b=*/false);
        }
        cur = nxt;
    }

    // ---- epilogue: C/D layout — VGPR r: lanes 0-15 -> M=r, lanes 16-31 -> M=r+8 ----
    TOut* outB = Out + (long)b * bStrideOut;
    const int mbase = m0 + wave * 16 + ((lane >> 4) * 8);
    const int cbase = n0 + (lane & 15);
#pragma unroll
    for (int n = 0; n < 4; ++n) {
#pragma unroll
        for (int r = 0; r < 8; ++r) {
            const int m = mbase + r;
            const int c = cbase + n * 16;
            outB[(long)m * N + c] = (TOut)(acc[n][r] + bias[m]);
        }
    }
}

// ---------------------------------------------------------------------------
// scores[b][h][g][t] = (1/8) * sum_d q[b,h,d,t] * k[b,g,d,t]
// qkv layout: [b][h*192 + j][t], j: 0-63 q, 64-127 k, 128-191 v
// ---------------------------------------------------------------------------
__global__ __launch_bounds__(256) void scores_kernel(
    const bf16* __restrict__ qkv, float* __restrict__ scores)
{
    const int t  = blockIdx.x * blockDim.x + threadIdx.x;
    const int hg = blockIdx.y;
    const int h  = hg >> 4, g = hg & 15;
    const int b  = blockIdx.z;

    const bf16* qp = qkv + ((long)b * QKV_CH + h * 192)      * T_DIM + t;
    const bf16* kp = qkv + ((long)b * QKV_CH + g * 192 + 64) * T_DIM + t;

    float acc = 0.0f;
#pragma unroll 8
    for (int d = 0; d < 64; ++d)
        acc += (float)qp[(long)d * T_DIM] * (float)kp[(long)d * T_DIM];

    scores[(((long)b * NH + h) * NH + g) * T_DIM + t] = acc * 0.125f;
}

// ---------------------------------------------------------------------------
// In-place softmax over the T axis, one block per (b,h,g) row.
// ---------------------------------------------------------------------------
__global__ __launch_bounds__(256) void softmax_t(float* __restrict__ s)
{
    __shared__ float red[256];
    float* p = s + (long)blockIdx.x * T_DIM;
    const int tid = threadIdx.x;

    float mx = -3.0e38f;
    for (int t = tid; t < T_DIM; t += 256) mx = fmaxf(mx, p[t]);
    red[tid] = mx; __syncthreads();
    for (int off = 128; off > 0; off >>= 1) {
        if (tid < off) red[tid] = fmaxf(red[tid], red[tid + off]);
        __syncthreads();
    }
    mx = red[0]; __syncthreads();

    float sum = 0.0f;
    for (int t = tid; t < T_DIM; t += 256) {
        float e = __expf(p[t] - mx);
        p[t] = e;
        sum += e;
    }
    red[tid] = sum; __syncthreads();
    for (int off = 128; off > 0; off >>= 1) {
        if (tid < off) red[tid] += red[tid + off];
        __syncthreads();
    }
    const float inv = 1.0f / red[0];
    for (int t = tid; t < T_DIM; t += 256) p[t] *= inv;
}

// ---------------------------------------------------------------------------
// ctx[b][h*64+d][t] = sum_g attn[b][h][g][t] * v[b,g,d,t]
// ---------------------------------------------------------------------------
__global__ __launch_bounds__(256) void context_kernel(
    const float* __restrict__ attn, const bf16* __restrict__ qkv,
    bf16* __restrict__ ctx)
{
    const int t = blockIdx.x * blockDim.x + threadIdx.x;
    const int h = blockIdx.y;
    const int b = blockIdx.z;

    float a[16];
    const float* ap = attn + (((long)b * NH + h) * NH) * T_DIM + t;
#pragma unroll
    for (int g = 0; g < 16; ++g) a[g] = ap[(long)g * T_DIM];

    const bf16* vp = qkv + ((long)b * QKV_CH) * T_DIM + t;
    bf16* cp = ctx + ((long)b * C_DIM + h * 64) * T_DIM + t;

    for (int d = 0; d < 64; ++d) {
        float acc = 0.0f;
#pragma unroll
        for (int g = 0; g < 16; ++g)
            acc += a[g] * (float)vp[((long)(g * 192 + 128 + d)) * T_DIM];
        cp[(long)d * T_DIM] = (bf16)acc;
    }
}

// ---------------------------------------------------------------------------
// Launch
// ---------------------------------------------------------------------------
extern "C" void kernel_launch(void* const* d_in, const int* in_sizes, int n_in,
                              void* d_out, int out_size, void* d_ws, size_t ws_size,
                              hipStream_t stream)
{
    const float* x      = (const float*)d_in[0];  // (B, C, T)
    const float* w_qkv  = (const float*)d_in[1];  // (3072, C)
    const float* b_qkv  = (const float*)d_in[2];  // (3072,)
    const float* w_out  = (const float*)d_in[3];  // (C, 1024)
    const float* b_out  = (const float*)d_in[4];  // (C,)
    float*       out    = (float*)d_out;          // (B, C, T)

    // ---- carve workspace (256B aligned regions) ----
    char* ws = (char*)d_ws;
    size_t off = 0;
    auto carve = [&](size_t bytes) -> char* {
        char* p = ws + off;
        off += (bytes + 255) & ~(size_t)255;
        return p;
    };
    const long nX    = (long)B_DIM * C_DIM * T_DIM;        // 33.5M
    const long nWq   = (long)QKV_CH * C_DIM;               // 3.1M
    const long nWo   = (long)C_DIM * C_DIM;                // 1.0M
    const long nQKV  = (long)B_DIM * QKV_CH * T_DIM;       // 100.6M
    const long nScr  = (long)B_DIM * NH * NH * T_DIM;      // 8.4M
    const long nCtx  = (long)B_DIM * C_DIM * T_DIM;        // 33.5M

    bf16*  xb     = (bf16*)carve(nX   * sizeof(bf16));
    bf16*  wqkvb  = (bf16*)carve(nWq  * sizeof(bf16));
    bf16*  woutb  = (bf16*)carve(nWo  * sizeof(bf16));
    bf16*  qkvb   = (bf16*)carve(nQKV * sizeof(bf16));
    float* scores = (float*)carve(nScr * sizeof(float));
    bf16*  ctxb   = (bf16*)carve(nCtx * sizeof(bf16));
    (void)ws_size; (void)in_sizes; (void)n_in; (void)out_size;

    // ---- 1. convert inputs to bf16 ----
    cvt_f32_bf16<<<4096, 256, 0, stream>>>(x,     xb,    nX);
    cvt_f32_bf16<<<1024, 256, 0, stream>>>(w_qkv, wqkvb, nWq);
    cvt_f32_bf16<<<512,  256, 0, stream>>>(w_out, woutb, nWo);

    // ---- 2. QKV projection: qkv = w_qkv @ x + b_qkv  (bf16 out) ----
    {
        dim3 grid(T_DIM / 64, QKV_CH / 128, B_DIM);
        gemm_bf16_wmma<bf16><<<grid, 256, 0, stream>>>(
            wqkvb, xb, b_qkv, qkvb,
            QKV_CH, C_DIM, T_DIM,
            (long)C_DIM * T_DIM, (long)QKV_CH * T_DIM);
    }

    // ---- 3. head-vs-head scores (with q scale folded in) ----
    scores_kernel<<<dim3(T_DIM / 256, NH * NH, B_DIM), 256, 0, stream>>>(qkvb, scores);

    // ---- 4. softmax over t (in place) ----
    softmax_t<<<B_DIM * NH * NH, 256, 0, stream>>>(scores);

    // ---- 5. context = attn-weighted head mix of V (bf16 out) ----
    context_kernel<<<dim3(T_DIM / 256, NH, B_DIM), 256, 0, stream>>>(scores, qkvb, ctxb);

    // ---- 6. output projection: out = w_out @ ctx + b_out  (f32 out) ----
    {
        dim3 grid(T_DIM / 64, C_DIM / 128, B_DIM);
        gemm_bf16_wmma<float><<<grid, 256, 0, stream>>>(
            woutb, ctxb, b_out, out,
            C_DIM, C_DIM, T_DIM,
            (long)C_DIM * T_DIM, (long)C_DIM * T_DIM);
    }
}
